// WindowAttention_4363686772926
// MI455X (gfx1250) — matrix operand order repaired and checked by reading, not compiled, not run on hardware
//
#include <hip/hip_runtime.h>
#include <hip/hip_bf16.h>
#include <stdint.h>

// ---------------- problem constants ----------------
#define B_TOT   4096
#define NTOK    49
#define CDIM    256
#define NH      4
#define HD      64
#define NPAD    64
#define NWIN    64
#define QK_SCALE 0.125f   // 64^-0.5

typedef __attribute__((ext_vector_type(16))) __bf16 v16bf;
typedef __attribute__((ext_vector_type(8)))  float  v8f;

union FragBF {
    v16bf v;
    uint32_t u[8];
    uint4 q2[2];
};

// round-to-nearest-even fp32 -> bf16 (bit pattern as ushort)
__device__ __forceinline__ uint16_t f2bf(float f) {
    uint32_t u = __builtin_bit_cast(uint32_t, f);
    u += 0x7FFFu + ((u >> 16) & 1u);
    return (uint16_t)(u >> 16);
}

// A-fragment (16x32 bf16): lane l holds row m=l%16; lane-half selects K group.
// VGPR0..7 <- K = k0 + half*16 + {0..15}
__device__ __forceinline__ void load_a_frag(FragBF& f, const uint16_t* base,
                                            int row, int stride, int k0, int half) {
    const uint16_t* p = base + row * stride + k0 + half * 16;
    f.q2[0] = *(const uint4*)(p);
    f.q2[1] = *(const uint4*)(p + 8);
}

// B-fragment (32x16 bf16): lane l holds column n=l%16; lane-half selects K 0-15 / 16-31.
// per lane: 16 consecutive bf16 along K.
__device__ __forceinline__ void load_b_frag(FragBF& f, const uint16_t* p) {
    f.q2[0] = *(const uint4*)(p);
    f.q2[1] = *(const uint4*)(p + 8);
}

// CDNA5 async copy: global -> LDS, 16B per lane, tracked by ASYNCcnt.
__device__ __forceinline__ void async_copy_b128(uint32_t lds_off, const void* gaddr) {
    asm volatile("global_load_async_to_lds_b128 %0, %1, off"
                 :: "v"(lds_off), "v"((uint64_t)(uintptr_t)gaddr)
                 : "memory");
}
__device__ __forceinline__ void async_wait0() {
    asm volatile("s_wait_asynccnt 0x0" ::: "memory");
}

// ---------------- prep: weight transpose->bf16 + bias gather ----------------
__global__ void prep_kernel(const float* __restrict__ qkv_w,   // [256][768]
                            const float* __restrict__ proj_w,  // [256][256]
                            const float* __restrict__ bias_table, // [169][4]
                            const int*   __restrict__ rel_index,  // [49*49]
                            uint16_t* __restrict__ qkv_wT,     // [768][256] bf16
                            uint16_t* __restrict__ proj_wT,    // [256][256] bf16
                            float*    __restrict__ rel_bias) { // [4][49][49]
    int i = blockIdx.x * blockDim.x + threadIdx.x;
    if (i < 768 * 256) {
        int n = i >> 8, k = i & 255;
        qkv_wT[i] = f2bf(qkv_w[(size_t)k * 768 + n]);
    }
    if (i < 256 * 256) {
        int n = i >> 8, k = i & 255;
        proj_wT[i] = f2bf(proj_w[(size_t)k * 256 + n]);
    }
    if (i < NH * NTOK * NTOK) {
        int h = i / (NTOK * NTOK);
        int nm = i % (NTOK * NTOK);
        rel_bias[i] = bias_table[rel_index[nm] * NH + h];
    }
}

// ---------------- fused per-window QKV + attention ----------------
__launch_bounds__(256)
__global__ void winattn_fused(const float* __restrict__ x,       // [4096][49][256]
                              const float* __restrict__ qkv_b,   // [768]
                              const float* __restrict__ mask,    // [64][49][49]
                              const uint16_t* __restrict__ qkv_wT, // [768][256] bf16
                              const float* __restrict__ rel_bias,  // [4][49][49]
                              uint16_t* __restrict__ attn_out) {   // [4096*49][256] bf16
    __shared__ __align__(16) uint16_t lds_x[NPAD * CDIM];  // 32 KB
    __shared__ __align__(16) uint16_t lds_q[NPAD * HD];    // 8 KB (pre-scaled q)
    __shared__ __align__(16) uint16_t lds_k[NPAD * HD];    // 8 KB
    __shared__ __align__(16) uint16_t lds_vT[HD * NPAD];   // 8 KB (v transposed [d][m])
    __shared__ __align__(16) uint16_t lds_p[NPAD * NPAD];  // 8 KB (softmax probs)

    const int b    = blockIdx.x;
    const int tid  = threadIdx.x;
    const int lane = tid & 31;
    const int wv   = tid >> 5;
    const int ln16 = lane & 15;
    const int half = lane >> 4;

    // ---- stage x window: 49x256 fp32 -> 64x256 bf16 (zero padded rows) ----
    const float* xb = x + (size_t)b * NTOK * CDIM;
    for (int i = tid; i < (NPAD * CDIM) / 4; i += 256) {
        int e = i * 4;
        int row = e >> 8;
        uint32_t p0 = 0u, p1 = 0u;
        if (row < NTOK) {
            float4 v = *(const float4*)(xb + (size_t)row * CDIM + (e & 255));
            p0 = (uint32_t)f2bf(v.x) | ((uint32_t)f2bf(v.y) << 16);
            p1 = (uint32_t)f2bf(v.z) | ((uint32_t)f2bf(v.w) << 16);
        }
        *(uint32_t*)&lds_x[e]     = p0;
        *(uint32_t*)&lds_x[e + 2] = p1;
    }
    __syncthreads();

    const float* maskw = mask + (size_t)(b & (NWIN - 1)) * NTOK * NTOK;

    for (int h = 0; h < NH; ++h) {
        // ======== per-head QKV: (64x256) @ (256 x 3*64), bf16 WMMA ========
        // 12 column tiles (q0..3, k0..3, v0..3); each wave owns a column tile
        // across ALL 4 m-tiles so every B-fragment feeds 4 WMMAs (4x less L2).
        for (int cc = 0; cc < 2; ++cc) {
            int ct = wv + cc * 8;
            if (ct < 12) {
                int mat = ct >> 2;             // 0=q 1=k 2=v
                int nt  = ct & 3;
                int colg = mat * CDIM + h * HD + nt * 16;   // global qkv out column
                const uint16_t* wrow = qkv_wT + (size_t)(colg + ln16) * CDIM;
                v8f acc[4] = {};
                #pragma unroll
                for (int ks = 0; ks < 8; ++ks) {
                    FragBF bm;
                    const uint16_t* bp = wrow + ks * 32 + half * 16;
                    __builtin_prefetch(bp + 32, 0, 1);      // next K slice
                    load_b_frag(bm, bp);
                    #pragma unroll
                    for (int mt = 0; mt < 4; ++mt) {
                        FragBF a;
                        load_a_frag(a, lds_x, mt * 16 + ln16, CDIM, ks * 32, half);
                        acc[mt] = __builtin_amdgcn_wmma_f32_16x16x32_bf16(
                                      false, a.v, false, bm.v, (short)0,
                                      acc[mt], false, false);
                    }
                }
                float biasc = qkv_b[colg + ln16];
                int d = nt * 16 + ln16;                      // head-local dim
                #pragma unroll
                for (int mt = 0; mt < 4; ++mt)
                    #pragma unroll
                    for (int r = 0; r < 8; ++r) {
                        int m = mt * 16 + r + half * 8;
                        float vv = acc[mt][r] + biasc;
                        if (mat == 0)      lds_q[m * HD + d]    = f2bf(vv * QK_SCALE);
                        else if (mat == 1) lds_k[m * HD + d]    = f2bf(vv);
                        else               lds_vT[d * NPAD + m] = f2bf(vv);
                    }
            }
        }
        __syncthreads();

        // ======== scores S = q @ k^T (64x64), bias+mask, softmax ========
        if (wv < 4) {
            int mt = wv;                                    // row tile
            v8f s[4] = {};
            #pragma unroll
            for (int nt = 0; nt < 4; ++nt) {
                #pragma unroll
                for (int ks = 0; ks < 2; ++ks) {
                    FragBF a, bm;
                    load_a_frag(a, lds_q, mt * 16 + ln16, HD, ks * 32, half);
                    load_b_frag(bm, lds_k + (nt * 16 + ln16) * HD + ks * 32 + half * 16);
                    s[nt] = __builtin_amdgcn_wmma_f32_16x16x32_bf16(
                                false, a.v, false, bm.v, (short)0, s[nt], false, false);
                }
            }
            // add relative-position bias + window mask; -inf on padded columns
            #pragma unroll
            for (int nt = 0; nt < 4; ++nt) {
                int col = nt * 16 + ln16;
                #pragma unroll
                for (int r = 0; r < 8; ++r) {
                    int m = mt * 16 + r + half * 8;
                    float add;
                    if (col >= NTOK)     add = -1e30f;
                    else if (m < NTOK)   add = rel_bias[((size_t)h * NTOK + m) * NTOK + col]
                                             + maskw[(size_t)m * NTOK + col];
                    else                 add = 0.0f;
                    s[nt][r] += add;
                }
            }
            // row softmax: reduce over 4 col tiles + 16 lanes of the half-wave
            float rmax[8], rsum[8];
            #pragma unroll
            for (int r = 0; r < 8; ++r) {
                float mx = fmaxf(fmaxf(s[0][r], s[1][r]), fmaxf(s[2][r], s[3][r]));
                #pragma unroll
                for (int sh = 1; sh < 16; sh <<= 1)
                    mx = fmaxf(mx, __shfl_xor(mx, sh, 32));
                rmax[r] = mx;
            }
            #pragma unroll
            for (int nt = 0; nt < 4; ++nt)
                #pragma unroll
                for (int r = 0; r < 8; ++r)
                    s[nt][r] = __expf(s[nt][r] - rmax[r]);
            #pragma unroll
            for (int r = 0; r < 8; ++r) {
                float sm = s[0][r] + s[1][r] + s[2][r] + s[3][r];
                #pragma unroll
                for (int sh = 1; sh < 16; sh <<= 1)
                    sm += __shfl_xor(sm, sh, 32);
                rsum[r] = 1.0f / sm;
            }
            #pragma unroll
            for (int nt = 0; nt < 4; ++nt)
                #pragma unroll
                for (int r = 0; r < 8; ++r)
                    lds_p[(mt * 16 + r + half * 8) * NPAD + nt * 16 + ln16] =
                        f2bf(s[nt][r] * rsum[r]);
        }
        __syncthreads();

        // ======== out = P @ V (64x64); store bf16 to global ========
        uint16_t* aout = attn_out + (size_t)b * NTOK * CDIM + h * HD;
        #pragma unroll
        for (int tt = 0; tt < 2; ++tt) {
            int tile = wv * 2 + tt;          // 0..15
            int mt = tile >> 2, dt = tile & 3;
            v8f acc = {};
            #pragma unroll
            for (int ks = 0; ks < 2; ++ks) {
                FragBF a, bm;
                load_a_frag(a, lds_p, mt * 16 + ln16, NPAD, ks * 32, half);
                load_b_frag(bm, lds_vT + (dt * 16 + ln16) * NPAD + ks * 32 + half * 16);
                acc = __builtin_amdgcn_wmma_f32_16x16x32_bf16(
                          false, a.v, false, bm.v, (short)0, acc, false, false);
            }
            #pragma unroll
            for (int r = 0; r < 8; ++r) {
                int m = mt * 16 + r + half * 8;
                if (m < NTOK)
                    aout[(size_t)m * CDIM + dt * 16 + ln16] = f2bf(acc[r]);
            }
        }
        __syncthreads();   // before next head reuses LDS
    }
}

// ---------------- output projection: (200704x256) @ (256x256) + b ----------------
__launch_bounds__(256)
__global__ void proj_kernel(const uint16_t* __restrict__ ain,     // [200704][256] bf16
                            const uint16_t* __restrict__ proj_wT, // [256][256] bf16
                            const float*    __restrict__ proj_b,  // [256]
                            float*          __restrict__ out) {   // [200704][256] f32
    __shared__ __align__(16) uint16_t lds_a[64 * CDIM];  // 32 KB
    const int tid  = threadIdx.x;
    const int lane = tid & 31;
    const int wv   = tid >> 5;
    const int ln16 = lane & 15;
    const int half = lane >> 4;
    const size_t rowbase = (size_t)blockIdx.x * 64;

    // stage 64x256 bf16 A tile (32 KB) via CDNA5 async global->LDS copies
    {
        const uint4* src = (const uint4*)(ain + rowbase * CDIM);
        for (int i = tid; i < 2048; i += 256) {
            uint32_t ldsoff = (uint32_t)(uintptr_t)(&((uint4*)lds_a)[i]);
            async_copy_b128(ldsoff, src + i);
        }
        async_wait0();
    }
    __syncthreads();

    const int n0 = wv * 2;                     // 2 column tiles per wave
    v8f acc[2][4] = {};
    #pragma unroll
    for (int ks = 0; ks < 8; ++ks) {
        FragBF a[4], bm[2];
        #pragma unroll
        for (int mt = 0; mt < 4; ++mt)
            load_a_frag(a[mt], lds_a, mt * 16 + ln16, CDIM, ks * 32, half);
        #pragma unroll
        for (int ni = 0; ni < 2; ++ni) {
            const uint16_t* bp = proj_wT
                + (size_t)((n0 + ni) * 16 + ln16) * CDIM + ks * 32 + half * 16;
            __builtin_prefetch(bp + 32, 0, 1);
            load_b_frag(bm[ni], bp);
        }
        #pragma unroll
        for (int ni = 0; ni < 2; ++ni)
            #pragma unroll
            for (int mt = 0; mt < 4; ++mt)
                acc[ni][mt] = __builtin_amdgcn_wmma_f32_16x16x32_bf16(
                                  false, a[mt].v, false, bm[ni].v, (short)0,
                                  acc[ni][mt], false, false);
    }
    #pragma unroll
    for (int ni = 0; ni < 2; ++ni) {
        int col = (n0 + ni) * 16 + ln16;
        float pb = proj_b[col];
        #pragma unroll
        for (int mt = 0; mt < 4; ++mt)
            #pragma unroll
            for (int r = 0; r < 8; ++r) {
                size_t m = rowbase + mt * 16 + r + half * 8;
                out[m * CDIM + col] = acc[ni][mt][r] + pb;
            }
    }
}

// ---------------- launch ----------------
extern "C" void kernel_launch(void* const* d_in, const int* in_sizes, int n_in,
                              void* d_out, int out_size, void* d_ws, size_t ws_size,
                              hipStream_t stream) {
    const float* x          = (const float*)d_in[0];
    const float* mask       = (const float*)d_in[1];
    const float* qkv_w      = (const float*)d_in[2];
    const float* qkv_b      = (const float*)d_in[3];
    const float* proj_w     = (const float*)d_in[4];
    const float* proj_b     = (const float*)d_in[5];
    const float* bias_table = (const float*)d_in[6];
    const int*   rel_index  = (const int*)d_in[7];
    float* out = (float*)d_out;

    char* ws = (char*)d_ws;
    uint16_t* qkv_wT   = (uint16_t*)(ws);            // 768*256*2  = 393216 B
    uint16_t* proj_wT  = (uint16_t*)(ws + 393216);   // 256*256*2  = 131072 B
    float*    rel_bias = (float*)   (ws + 524288);   // 4*49*49*4  =  38416 B
    uint16_t* attn_out = (uint16_t*)(ws + 589824);   // 200704*256*2 = 102760448 B

    prep_kernel<<<768, 256, 0, stream>>>(qkv_w, proj_w, bias_table, rel_index,
                                         qkv_wT, proj_wT, rel_bias);
    winattn_fused<<<B_TOT, 256, 0, stream>>>(x, qkv_b, mask, qkv_wT, rel_bias, attn_out);
    proj_kernel<<<(B_TOT * NTOK) / 64, 256, 0, stream>>>(attn_out, proj_wT, proj_b, out);
}
